// MoELayer_71777493451378
// MI455X (gfx1250) — matrix operand order repaired and checked by reading, not compiled, not run on hardware
//
#include <hip/hip_runtime.h>
#include <hip/hip_bf16.h>
#include <math.h>

typedef unsigned short u16;
typedef __attribute__((ext_vector_type(8)))  u16    u16x8;
typedef __attribute__((ext_vector_type(16))) u16    u16x16;
typedef __attribute__((ext_vector_type(16))) __bf16 v16bf;
typedef __attribute__((ext_vector_type(8)))  float  v8f;
typedef __attribute__((ext_vector_type(4)))  unsigned int u32x4;
typedef __attribute__((ext_vector_type(8)))  int    i32x8;
typedef __attribute__((ext_vector_type(4)))  int    i32x4;

constexpr int T = 4096, H = 1024, F = 2048, E = 8, TOPK = 2;

#if defined(__has_builtin)
#if __has_builtin(__builtin_amdgcn_tensor_load_to_lds) && __has_builtin(__builtin_amdgcn_s_wait_tensorcnt)
#define USE_TDM 1
#endif
#endif

__device__ inline u16 f2bf(float f) {
  unsigned u = __builtin_bit_cast(unsigned, f);
  u += 0x7FFFu + ((u >> 16) & 1u);          // round-to-nearest-even
  return (u16)(u >> 16);
}

// two fp32 -> packed bf16 dword (lo = a, hi = b)
__device__ inline unsigned pack_bf16(float a, float b) {
#if defined(__has_builtin) && __has_builtin(__builtin_amdgcn_cvt_pk_bf16_f32)
  auto r = __builtin_amdgcn_cvt_pk_bf16_f32(a, b);
  unsigned u;
  __builtin_memcpy(&u, &r, 4);
  return u;
#else
  __hip_bfloat162 r = __float22bfloat162_rn(make_float2(a, b));
  unsigned u;
  __builtin_memcpy(&u, &r, 4);
  return u;
#endif
}

__device__ inline float gelu_tanh(float x) {
  float x3 = x * x * x;
  return 0.5f * x * (1.0f + tanhf(0.7978845608028654f * (x + 0.044715f * x3)));
}

__device__ inline unsigned ldsoff(const void* p) {
  return (unsigned)(unsigned long long)(uintptr_t)p;
}

#ifdef USE_TDM
// One TDM 2D tile load: tile_rows x 32 bf16 elements, row stride kt elements,
// into LDS at lds_off with 16B pad after every 64B row (stride-40 u16 layout).
// Rows >= dim1 are hardware zero-filled (tensor_dim1 OOB -> 0).
__device__ inline void tdm_load(const u16* gsrc, unsigned lds_off, int dim1,
                                int tile_rows, int kt) {
  unsigned long long ga = (unsigned long long)(uintptr_t)gsrc;
  u32x4 g0;
  g0[0] = 1u;                                                // count=1, user mode
  g0[1] = lds_off;                                           // LDS byte address
  g0[2] = (unsigned)(ga & 0xFFFFFFFFu);                      // global_addr[31:0]
  g0[3] = (unsigned)((ga >> 32) & 0x1FFFFFFu) | (2u << 30);  // addr[56:32] | type=2
  i32x8 g1;
  g1[0] = (1 << 16)    // data_size = 2 bytes
        | (1 << 20)    // pad_enable
        | (3 << 22)    // pad_interval: 16 dwords (one 64B row)
        | (3 << 25);   // pad_amount: 4 dwords (16B)
  g1[1] = (32 << 16);                  // tensor_dim0[15:0]=32 at bits[63:48]
  g1[2] = (dim1 & 0xFFFF) << 16;       // tensor_dim1[15:0]
  g1[3] = (32 << 16);                  // tile_dim0=32 at bits[127:112]
  g1[4] = tile_rows & 0xFFFF;          // tile_dim1, tile_dim2=0
  g1[5] = kt;                          // tensor_dim0_stride[31:0] (elements)
  g1[6] = 0;
  g1[7] = 0;
  i32x4 z4 = {0, 0, 0, 0};
#if __has_include(<hip/amd_detail/amd_gfx1250_TDM.h>)
  i32x8 z8 = {0, 0, 0, 0, 0, 0, 0, 0};
  __builtin_amdgcn_tensor_load_to_lds(g0, g1, z4, z4, z8, 0);
#else
  __builtin_amdgcn_tensor_load_to_lds(g0, g1, z4, z4, 0);
#endif
}
#endif

// ---------------------------------------------------------------- init ------
__global__ void moe_init_kernel(float* __restrict__ out, int n4, int* __restrict__ counts) {
  int i = blockIdx.x * blockDim.x + threadIdx.x;
  if (i < n4) *(float4*)(out + (size_t)i * 4) = make_float4(0.f, 0.f, 0.f, 0.f);
  if (i < E) counts[i] = 0;
}

// ------------------------------------------------- weight convert+transpose -
// src: per-expert [K][N] fp32  ->  dst: per-expert [N][K] bf16 (32x32 LDS tiles)
template <int K, int N>
__global__ __launch_bounds__(256) void moe_wconv_kernel(const float* __restrict__ src,
                                                        u16* __restrict__ dst) {
  __shared__ u16 tile[32][34];
  const float* S = src + (size_t)blockIdx.z * K * N;
  u16*         D = dst + (size_t)blockIdx.z * K * N;
  const int k0 = blockIdx.y * 32, n0 = blockIdx.x * 32;
  {
    const int tr = threadIdx.x >> 3;          // k within tile
    const int tc = (threadIdx.x & 7) * 4;     // n chunk
    float4 v = *(const float4*)(S + (size_t)(k0 + tr) * N + n0 + tc);
    tile[tc + 0][tr] = f2bf(v.x);
    tile[tc + 1][tr] = f2bf(v.y);
    tile[tc + 2][tr] = f2bf(v.z);
    tile[tc + 3][tr] = f2bf(v.w);
  }
  __syncthreads();
  {
    const int orow = threadIdx.x >> 3;        // n within tile
    const int oc = (threadIdx.x & 7) * 4;     // k chunk
    unsigned lo = *(const unsigned*)&tile[orow][oc];
    unsigned hi = *(const unsigned*)&tile[orow][oc + 2];
    uint2 o; o.x = lo; o.y = hi;
    *(uint2*)(D + (size_t)(n0 + orow) * K + k0 + oc) = o;
  }
}

// -------------------------------------------------------------- router ------
__global__ void moe_router_kernel(const float* __restrict__ hs, const float* __restrict__ rw,
                                  int* __restrict__ t2i, float* __restrict__ t2p,
                                  int* __restrict__ counts) {
  const int wave = threadIdx.x >> 5, lane = threadIdx.x & 31;
  const int t = blockIdx.x * 8 + wave;
  float acc[E];
#pragma unroll
  for (int e = 0; e < E; ++e) acc[e] = 0.f;
  const float* xr = hs + (size_t)t * H;
  for (int h = lane; h < H; h += 32) {
    float x = xr[h];
    const float4* r4 = (const float4*)(rw + (size_t)h * E);
    float4 a = r4[0], b = r4[1];
    acc[0] += x * a.x; acc[1] += x * a.y; acc[2] += x * a.z; acc[3] += x * a.w;
    acc[4] += x * b.x; acc[5] += x * b.y; acc[6] += x * b.z; acc[7] += x * b.w;
  }
#pragma unroll
  for (int e = 0; e < E; ++e)
#pragma unroll
    for (int s = 16; s > 0; s >>= 1) acc[e] += __shfl_xor(acc[e], s, 32);
  if (lane == 0) {
    float v0 = -3.0e38f; int i0 = 0;
#pragma unroll
    for (int e = 0; e < E; ++e) if (acc[e] > v0) { v0 = acc[e]; i0 = e; }
    float v1 = -3.0e38f; int i1 = 0;
#pragma unroll
    for (int e = 0; e < E; ++e) if (e != i0 && acc[e] > v1) { v1 = acc[e]; i1 = e; }
    float p0 = 1.f / (1.f + __expf(v1 - v0));
    t2i[t * 2 + 0] = i0; t2i[t * 2 + 1] = i1;
    t2p[t * 2 + 0] = p0; t2p[t * 2 + 1] = 1.f - p0;
    atomicAdd(&counts[i0], 1);
    atomicAdd(&counts[i1], 1);
  }
}

// ---------------------------------------------------------------- scan ------
__global__ void moe_scan_kernel(const int* __restrict__ counts,
                                int* __restrict__ offsets, int* __restrict__ cursors) {
  if (threadIdx.x == 0) {
    int o = 0;
#pragma unroll
    for (int e = 0; e < E; ++e) { offsets[e] = o; cursors[e] = o; o += counts[e]; }
    offsets[E] = o;   // == 2T
  }
}

// -------------------------------------------------------------- gather ------
__global__ void moe_gather_kernel(const float* __restrict__ hs,
                                  const int* __restrict__ t2i, const float* __restrict__ t2p,
                                  int* __restrict__ cursors,
                                  int* __restrict__ rowtok, float* __restrict__ rowprob,
                                  u16* __restrict__ Xg) {
  __shared__ int rsh[TOPK];
  const int t = blockIdx.x;
  if (threadIdx.x == 0) {
#pragma unroll
    for (int k = 0; k < TOPK; ++k) {
      int e = t2i[t * 2 + k];
      int r = atomicAdd(&cursors[e], 1);
      rsh[k] = r;
      rowtok[r] = t;
      rowprob[r] = t2p[t * 2 + k];
    }
  }
  __syncthreads();
  const int h = threadIdx.x * 4;
  float4 v = *(const float4*)(hs + (size_t)t * H + h);
  uint2 b;
  b.x = pack_bf16(v.x, v.y);
  b.y = pack_bf16(v.z, v.w);
#pragma unroll
  for (int k = 0; k < TOPK; ++k)
    *(uint2*)(Xg + (size_t)rsh[k] * H + h) = b;
}

// ---------------------------------------------------------------- gemm ------
// Block tile 128(M) x 128(N), K-step 32; 8 waves in 2(M)x4(N), wave tile 64x32
// -> eight v_wmma_f32_16x16x32_bf16 per wave per K-step (1.5 ds_loads/WMMA).
// A and B(T) tiles are identical 128-row x 32-elem bf16 tiles, double-buffered
// via TDM tensor_load_to_lds; one barrier + s_wait_tensorcnt per K-step.
template <int KT, int NT, int EPI>
__global__ __launch_bounds__(256) void moe_gemm_kernel(
    const u16* __restrict__ Ag, const u16* __restrict__ BT,   // BT: [E][NT][KT] bf16
    u16* __restrict__ Cbf, float* __restrict__ out,
    const int* __restrict__ offsets,
    const int* __restrict__ rowtok, const float* __restrict__ rowprob) {
  const int e   = blockIdx.z;
  const int off = offsets[e];
  const int ne  = offsets[e + 1] - off;
  const int m0  = blockIdx.x * 128;
  if (m0 >= ne) return;
  const int n0  = blockIdx.y * 128;

  __shared__ __align__(16) u16 shA[2][128 * 40];   // [M][K] bf16, 16B row pad
  __shared__ __align__(16) u16 shB[2][128 * 40];   // [N][K] bf16, 16B row pad

  const int tid  = threadIdx.x;
  const int lane = tid & 31, wave = tid >> 5;
  const int l16  = lane & 15, lh = lane >> 4;
  const int wm   = wave & 1,  wn = wave >> 1;      // wave tile: 64(M) x 32(N)

  const u16* abase = Ag + (size_t)(off + m0) * KT;
  const u16* bbase = BT + (size_t)e * NT * KT + (size_t)n0 * KT;
  const int rows_avail = ne - m0;                  // TDM zero-fills rows >= this

  v8f acc[4][2] = {};

  auto compute_tile = [&](const u16* bufA, const u16* bufB) {
    // A frag (16-bit 16x32): lane<16 -> K[0..8)+[16..24); lane>=16 -> K[8..16)+[24..32)
    v16bf aF[4], bF[2];
#pragma unroll
    for (int mi = 0; mi < 4; ++mi) {
      const u16* p = bufA + (wm * 64 + mi * 16 + l16) * 40 + lh * 8;
      u16x8 lo = *(const u16x8*)p;
      u16x8 hi = *(const u16x8*)(p + 16);
      u16x16 f = __builtin_shufflevector(lo, hi, 0, 1, 2, 3, 4, 5, 6, 7,
                                         8, 9, 10, 11, 12, 13, 14, 15);
      aF[mi] = __builtin_bit_cast(v16bf, f);
    }
    // B frag (32x16): lane<16 -> K[0..8)+[16..24) of col lane; lane>=16 -> K[8..16)+[24..32)
#pragma unroll
    for (int ni = 0; ni < 2; ++ni) {
      const u16* p = bufB + (wn * 32 + ni * 16 + l16) * 40 + lh * 8;
      u16x8 lo = *(const u16x8*)p;
      u16x8 hi = *(const u16x8*)(p + 16);
      u16x16 f = __builtin_shufflevector(lo, hi, 0, 1, 2, 3, 4, 5, 6, 7,
                                         8, 9, 10, 11, 12, 13, 14, 15);
      bF[ni] = __builtin_bit_cast(v16bf, f);
    }
#pragma unroll
    for (int mi = 0; mi < 4; ++mi)
#pragma unroll
      for (int ni = 0; ni < 2; ++ni)
        acc[mi][ni] = __builtin_amdgcn_wmma_f32_16x16x32_bf16(
            false, aF[mi], false, bF[ni], (short)0, acc[mi][ni], false, false);
  };

#ifdef USE_TDM
  constexpr int ITER = KT / 32;
  if (wave == 0) {
    tdm_load(abase, ldsoff(&shA[0][0]), rows_avail, 128, KT);
    tdm_load(bbase, ldsoff(&shB[0][0]), 128, 128, KT);
  }
  for (int i = 0; i < ITER; ++i) {
    const int buf = i & 1;
    if (wave == 0) __builtin_amdgcn_s_wait_tensorcnt(0);  // pair i landed
    __syncthreads();                                      // all waves see buf i
    if (wave == 0 && i + 1 < ITER) {                      // prefetch pair i+1
      tdm_load(abase + (i + 1) * 32, ldsoff(&shA[buf ^ 1][0]), rows_avail, 128, KT);
      tdm_load(bbase + (i + 1) * 32, ldsoff(&shB[buf ^ 1][0]), 128, 128, KT);
    }
    compute_tile(&shA[buf][0], &shB[buf][0]);
    // next-iteration wait+barrier doubles as the release barrier for buf
  }
#else
  const int arow = tid >> 1, achunk = (tid & 1) * 16;
  const int agr  = min(m0 + arow, ne - 1);
  const u16* aptr = abase + (size_t)(min(arow, ne - 1 - m0 >= 0 ? arow : 0)) * 0 +
                    (size_t)(agr - m0) * KT + achunk;    // clamped row
  const int brow = tid >> 1, bc = (tid & 1) * 16;
  const u16* bptr = bbase + (size_t)brow * KT + bc;
  for (int k0 = 0; k0 < KT; k0 += 32) {
    *(uint4*)&shA[0][arow * 40 + achunk]     = *(const uint4*)(aptr + k0);
    *(uint4*)&shA[0][arow * 40 + achunk + 8] = *(const uint4*)(aptr + k0 + 8);
    *(uint4*)&shB[0][brow * 40 + bc]         = *(const uint4*)(bptr + k0);
    *(uint4*)&shB[0][brow * 40 + bc + 8]     = *(const uint4*)(bptr + k0 + 8);
    __syncthreads();
    compute_tile(&shA[0][0], &shB[0][0]);
    __syncthreads();
  }
#endif

  // C/D layout: element (v, lane): M = v + 8*(lane>=16), N = lane%16
#pragma unroll
  for (int mi = 0; mi < 4; ++mi) {
#pragma unroll
    for (int ni = 0; ni < 2; ++ni) {
      const int col = n0 + wn * 32 + ni * 16 + l16;
#pragma unroll
      for (int v = 0; v < 8; ++v) {
        const int row = m0 + wm * 64 + mi * 16 + lh * 8 + v;
        if (row < ne) {
          float c = acc[mi][ni][v];
          if constexpr (EPI == 0) {
            Cbf[(size_t)(off + row) * NT + col] = f2bf(gelu_tanh(c));
          } else {
            int   t = rowtok[off + row];
            float p = rowprob[off + row];
            atomicAdd(&out[(size_t)t * NT + col], p * c);
          }
        }
      }
    }
  }
}

// ---------------------------------------------------------------------------
extern "C" void kernel_launch(void* const* d_in, const int* in_sizes, int n_in,
                              void* d_out, int out_size, void* d_ws, size_t ws_size,
                              hipStream_t stream) {
  (void)in_sizes; (void)n_in; (void)ws_size;
  const float* hs = (const float*)d_in[0];   // [T,H]
  const float* rw = (const float*)d_in[1];   // [H,E]
  const float* w1 = (const float*)d_in[2];   // [E,H,F]
  const float* w2 = (const float*)d_in[3];   // [E,F,H]
  float* out = (float*)d_out;                // [T,H]

  char* ws = (char*)d_ws;
  int*   counts  = (int*)(ws + 0);
  int*   offsets = (int*)(ws + 256);
  int*   cursors = (int*)(ws + 512);
  int*   t2i     = (int*)  (ws + 4096);
  float* t2p     = (float*)(ws + 36864);
  int*   rowtok  = (int*)  (ws + 69632);
  float* rowprob = (float*)(ws + 102400);
  constexpr size_t XG_OFF = 135168;
  constexpr size_t HG_OFF = XG_OFF + (size_t)2 * T * H * 2;   // +16 MB
  constexpr size_t W1_OFF = HG_OFF + (size_t)2 * T * F * 2;   // +32 MB
  constexpr size_t W2_OFF = W1_OFF + (size_t)E * H * F * 2;   // +33.5 MB
  u16* Xg  = (u16*)(ws + XG_OFF);   // [2T][H] bf16
  u16* Hg  = (u16*)(ws + HG_OFF);   // [2T][F] bf16
  u16* W1T = (u16*)(ws + W1_OFF);   // [E][F][H] bf16 (w1 transposed)
  u16* W2T = (u16*)(ws + W2_OFF);   // [E][H][F] bf16 (w2 transposed)

  {
    int n4 = out_size / 4;
    moe_init_kernel<<<(n4 + 255) / 256, 256, 0, stream>>>(out, n4, counts);
  }
  // one-time weight convert + transpose: fp32 [K][N] -> bf16 [N][K]
  moe_wconv_kernel<H, F><<<dim3(F / 32, H / 32, E), 256, 0, stream>>>(w1, W1T);
  moe_wconv_kernel<F, H><<<dim3(H / 32, F / 32, E), 256, 0, stream>>>(w2, W2T);

  moe_router_kernel<<<T / 8, 256, 0, stream>>>(hs, rw, t2i, t2p, counts);
  moe_scan_kernel<<<1, 32, 0, stream>>>(counts, offsets, cursors);
  moe_gather_kernel<<<T, 256, 0, stream>>>(hs, t2i, t2p, cursors, rowtok, rowprob, Xg);
  moe_gemm_kernel<H, F, 0><<<dim3(T / 128, F / 128, E), 256, 0, stream>>>(
      Xg, W1T, Hg, nullptr, offsets, nullptr, nullptr);
  moe_gemm_kernel<F, H, 1><<<dim3(T / 128, H / 128, E), 256, 0, stream>>>(
      Hg, W2T, nullptr, out, offsets, rowtok, rowprob);
}
